// Attention_8907762172416
// MI455X (gfx1250) — matrix operand order repaired and checked by reading, not compiled, not run on hardware
//
#include <hip/hip_runtime.h>
#include <stdint.h>

typedef __attribute__((ext_vector_type(16))) _Float16 v16h;
typedef __attribute__((ext_vector_type(8)))  _Float16 v8h;
typedef __attribute__((ext_vector_type(8)))  float    v8f;
typedef __attribute__((ext_vector_type(4)))  unsigned int u32x4;
typedef __attribute__((ext_vector_type(4)))  int      i32x4;
typedef __attribute__((ext_vector_type(8)))  int      i32x8;

#define B_ 4
#define T_ 4096
#define D_ 2048
#define H_ 128
#define SC_ 64   // keys per chunk in flash loop

__device__ __forceinline__ v16h cat8(v8h a, v8h b) {
  return __builtin_shufflevector(a, b, 0,1,2,3,4,5,6,7,8,9,10,11,12,13,14,15);
}

__device__ __forceinline__ v8f wmma_f16(v16h a, v16h b, v8f c) {
  return __builtin_amdgcn_wmma_f32_16x16x32_f16(false, a, false, b, (short)0, c, false, false);
}

// ---------------------------------------------------------------------------
// Kernel 1: W (D,H) fp32 -> Wt (H,D) f16 (B-fragment K-pairs contiguous).
// ---------------------------------------------------------------------------
__global__ void prep_w(const float* __restrict__ Wq, const float* __restrict__ Wk,
                       const float* __restrict__ Wv, _Float16* __restrict__ wt) {
  int idx = blockIdx.x * 256 + threadIdx.x;
  int mat = idx / (H_ * D_);
  int rem = idx - mat * (H_ * D_);
  int h   = rem / D_;
  int kk  = rem - h * D_;
  const float* W = (mat == 0) ? Wq : (mat == 1) ? Wk : Wv;
  wt[idx] = (_Float16)W[(size_t)kk * H_ + h];
}

// ---------------------------------------------------------------------------
// Kernel 2: fused QKV projection.  One block per 16-row x tile; 12 waves =
// 3 mats x 4 N-pairs.  The x tile is converted to f16 once into LDS per
// K-step and shared by all waves, so x is read from HBM exactly once.
// Q is pre-scaled by log2(e)/sqrt(H) so flash softmax can use raw v_exp_f32.
// ---------------------------------------------------------------------------
__global__ __launch_bounds__(384) void proj_qkv(
    const float* __restrict__ x, const _Float16* __restrict__ wt,
    _Float16* __restrict__ q, _Float16* __restrict__ k, _Float16* __restrict__ v) {
  __shared__ _Float16 xs[16 * 32];

  const int tid  = threadIdx.x;
  const int lane = tid & 31;
  const int wave = tid >> 5;          // 0..11
  const int mat  = wave >> 2;         // 0..2
  const int sub  = wave & 3;          // 0..3 -> N-tiles {2*sub, 2*sub+1}
  const int row0 = blockIdx.x << 4;

  const _Float16* w = wt + (size_t)mat * H_ * D_;
  _Float16* outp = (mat == 0) ? q : (mat == 1) ? k : v;
  const float sc = (mat == 0) ? 1.4426950408889634f * 0.08838834764831845f : 1.0f;

  const int lhi = lane >> 4;
  const int llo = lane & 15;
  const int h0  = sub * 32 + llo;
  const int h1  = h0 + 16;

  v8f acc0 = {}; v8f acc1 = {};

  for (int k0 = 0; k0 < D_; k0 += 32) {
    __syncthreads();                  // previous A-frag readers done
    if (tid < 128) {                  // stage 16x32 f32 tile -> f16 LDS
      int idx = tid << 2;
      int r  = idx >> 5;
      int cc = idx & 31;
      const float4 f = *(const float4*)(x + (size_t)(row0 + r) * D_ + k0 + cc);
      _Float16* d = &xs[r * 32 + cc];
      d[0] = (_Float16)f.x; d[1] = (_Float16)f.y;
      d[2] = (_Float16)f.z; d[3] = (_Float16)f.w;
    }
    __syncthreads();

    // A-fragment (16-bit A layout) from LDS
    const v8h* ap = (const v8h*)(&xs[llo * 32 + (lhi << 3)]);
    v16h A = cat8(ap[0], ap[2]);

    const v8h* w0 = (const v8h*)(w + (size_t)h0 * D_ + k0 + (lhi << 3));
    const v8h* w1 = (const v8h*)(w + (size_t)h1 * D_ + k0 + (lhi << 3));
    acc0 = wmma_f16(A, cat8(w0[0], w0[2]), acc0);
    acc1 = wmma_f16(A, cat8(w1[0], w1[2]), acc1);
  }

#pragma unroll
  for (int r = 0; r < 8; r++) {
    int orow = row0 + r + (lhi << 3);
    outp[(size_t)orow * H_ + h0] = (_Float16)(acc0[r] * sc);
    outp[(size_t)orow * H_ + h1] = (_Float16)(acc1[r] * sc);
  }
}

// ---------------------------------------------------------------------------
// Flash attention staging: K chunk (contiguous 16KB) via TDM descriptor,
// V chunk via per-lane async b128 copies.
// ---------------------------------------------------------------------------
__device__ __forceinline__ void stage_kv(int s0, uint32_t ks_lds, uint32_t vrow_lds,
                                         const _Float16* kbase, const _Float16* vbase,
                                         int wave, int tid) {
  if (wave == 0) {
    uint64_t ga = (uint64_t)(uintptr_t)kbase + (uint64_t)(uint32_t)s0 * (H_ * 2);
    u32x4 g0;
    g0[0] = 1u;                                   // count=1, user mode
    g0[1] = ks_lds;                               // lds_addr
    g0[2] = (uint32_t)ga;                         // global_addr[31:0]
    g0[3] = (uint32_t)(ga >> 32) | (2u << 30);    // global_addr[56:32], type=2
    i32x8 g1;
    g1[0] = (int)(1u << 16);                      // data_size = 2 bytes
    g1[1] = (int)((SC_ * H_) << 16);              // tensor_dim0[15:0] = 8192
    g1[2] = (int)(1u << 16);                      // tensor_dim0 hi=0, tensor_dim1=1
    g1[3] = (int)((SC_ * H_) << 16);              // tile_dim0 = 8192 elements
    g1[4] = 1;                                    // tile_dim1=1, tile_dim2=0
    g1[5] = SC_ * H_;                             // tensor_dim0_stride lo
    g1[6] = 0;
    g1[7] = 0;
    i32x4 gz = {0, 0, 0, 0};
#if __clang_major__ >= 23
    i32x8 gz8 = {0, 0, 0, 0, 0, 0, 0, 0};
    __builtin_amdgcn_tensor_load_to_lds(g0, g1, gz, gz, gz8, 0);
#else
    __builtin_amdgcn_tensor_load_to_lds(g0, g1, gz, gz, 0);
#endif
  }
#pragma unroll
  for (int it = 0; it < 8; it++) {
    uint32_t c16  = (uint32_t)(tid + it * 128) * 16u;
    uint32_t ldsa = vrow_lds + c16;
    uint32_t goff = (uint32_t)s0 * (H_ * 2) + c16;
    asm volatile("global_load_async_to_lds_b128 %0, %1, %2"
                 :: "v"(ldsa), "v"(goff), "s"(vbase) : "memory");
  }
}

// ---------------------------------------------------------------------------
// Kernel 3: flash attention, grid = (T/64, B), block = 128 (4 waves, one
// 16-row query tile each), 64-key chunks, double-buffered K/V staging so the
// TDM + async copies for chunk i+1 overlap the 32 WMMAs of chunk i.
// ---------------------------------------------------------------------------
__global__ __launch_bounds__(128) void flash_attn(
    const _Float16* __restrict__ q, const _Float16* __restrict__ k,
    const _Float16* __restrict__ v, float* __restrict__ out) {
  __shared__ _Float16 Ks[2][SC_ * H_];     // [s][h] row-major (TDM dest)
  __shared__ _Float16 Vrow[2][SC_ * H_];   // [s][h] row-major (async dest)
  __shared__ _Float16 Vs[H_ * SC_];        // [h][s] transposed for B-frags
  __shared__ _Float16 Pl[4][16 * SC_];     // per-wave P scratch (C->A relayout)

  const int tid  = threadIdx.x;
  const int lane = tid & 31;
  const int wave = tid >> 5;
  const int b    = blockIdx.y;
  const int q0   = blockIdx.x * 64 + wave * 16;
  const size_t bT = (size_t)b * T_;

  const int lhi = lane >> 4;
  const int llo = lane & 15;
  const int qrow = q0 + llo;

  // Q fragments (16-bit A layout)
  v16h Qf[4];
#pragma unroll
  for (int c = 0; c < 4; c++) {
    const v8h* qp = (const v8h*)(q + (bT + qrow) * H_ + c * 32 + (lhi << 3));
    Qf[c] = cat8(qp[0], qp[2]);
  }

  v8f O[8];
#pragma unroll
  for (int i = 0; i < 8; i++) O[i] = (v8f){};
  float m[8], l[8];
#pragma unroll
  for (int r = 0; r < 8; r++) { m[r] = -1e30f; l[r] = 0.0f; }

  _Float16* Pw = &Pl[wave][0];
  const uint32_t ks_lds[2]   = { (uint32_t)(uintptr_t)(void*)&Ks[0][0],
                                 (uint32_t)(uintptr_t)(void*)&Ks[1][0] };
  const uint32_t vrow_lds[2] = { (uint32_t)(uintptr_t)(void*)&Vrow[0][0],
                                 (uint32_t)(uintptr_t)(void*)&Vrow[1][0] };
  const _Float16* kbase = k + bT * H_;
  const _Float16* vbase = v + bT * H_;

  const int NIT = T_ / SC_;
  stage_kv(0, ks_lds[0], vrow_lds[0], kbase, vbase, wave, tid);   // prologue

  for (int i = 0; i < NIT; i++) {
    const int cur = i & 1;

    asm volatile("s_wait_asynccnt 0" ::: "memory");
    if (wave == 0) __builtin_amdgcn_s_wait_tensorcnt(0);
    __syncthreads();      // buffers ready; previous compute done with nxt

    if (i + 1 < NIT)      // overlap next chunk's DMA with this chunk's WMMAs
      stage_kv((i + 1) * SC_, ks_lds[cur ^ 1], vrow_lds[cur ^ 1],
               kbase, vbase, wave, tid);

    // transpose Vrow[cur] [s][h] -> Vs [h][s], packed 32-bit stores (s-pairs)
    for (int c = tid; c < 512; c += 128) {
      int sp = c >> 4;                 // s-pair index 0..31
      int h8 = (c & 15) << 3;          // 0..120
      v8h a = *(const v8h*)(&Vrow[cur][(2 * sp) * H_ + h8]);
      v8h bb = *(const v8h*)(&Vrow[cur][(2 * sp + 1) * H_ + h8]);
#pragma unroll
      for (int j = 0; j < 8; j++) {
        union { _Float16 h[2]; uint32_t u; } pk;
        pk.h[0] = a[j]; pk.h[1] = bb[j];
        *(uint32_t*)(&Vs[(h8 + j) * SC_ + 2 * sp]) = pk.u;
      }
    }
    __syncthreads();

    // ---- scores S (16x64) = Q @ K^T : 16 WMMAs
    v8f S[4];
#pragma unroll
    for (int j = 0; j < 4; j++) S[j] = (v8f){};
#pragma unroll
    for (int c = 0; c < 4; c++) {
      const int kb = c * 32 + (lhi << 3);
#pragma unroll
      for (int j = 0; j < 4; j++) {
        const v8h* kp = (const v8h*)(&Ks[cur][(j * 16 + llo) * H_ + kb]);
        S[j] = wmma_f16(Qf[c], cat8(kp[0], kp[2]), S[j]);
      }
    }

    // ---- online softmax (log2 domain)
#pragma unroll
    for (int r = 0; r < 8; r++) {
      float t = fmaxf(fmaxf(S[0][r], S[1][r]), fmaxf(S[2][r], S[3][r]));
      t = fmaxf(t, __shfl_xor(t, 1));
      t = fmaxf(t, __shfl_xor(t, 2));
      t = fmaxf(t, __shfl_xor(t, 4));
      t = fmaxf(t, __shfl_xor(t, 8));
      float mn = fmaxf(m[r], t);
      float al = __builtin_amdgcn_exp2f(m[r] - mn);
      m[r] = mn;
      float ps = 0.0f;
#pragma unroll
      for (int j = 0; j < 4; j++) {
        float p = __builtin_amdgcn_exp2f(S[j][r] - mn);
        Pw[(r + (lhi << 3)) * SC_ + j * 16 + llo] = (_Float16)p;
        ps += p;
      }
      ps += __shfl_xor(ps, 1);
      ps += __shfl_xor(ps, 2);
      ps += __shfl_xor(ps, 4);
      ps += __shfl_xor(ps, 8);
      l[r] = l[r] * al + ps;
#pragma unroll
      for (int nt = 0; nt < 8; nt++) O[nt][r] *= al;
    }

    // ---- O += P (16x64) @ V (64x128) : 16 WMMAs
#pragma unroll
    for (int kc = 0; kc < 2; kc++) {
      const v8h* pp = (const v8h*)(Pw + llo * SC_ + kc * 32 + (lhi << 3));
      v16h Pf = cat8(pp[0], pp[2]);
#pragma unroll
      for (int nt = 0; nt < 8; nt++) {
        const v8h* vp = (const v8h*)(&Vs[(nt * 16 + llo) * SC_ + kc * 32 + (lhi << 3)]);
        O[nt] = wmma_f16(Pf, cat8(vp[0], vp[2]), O[nt]);
      }
    }
  }

  // ---- epilogue: normalize by row sums, store fp32
#pragma unroll
  for (int nt = 0; nt < 8; nt++) {
#pragma unroll
    for (int r = 0; r < 8; r++) {
      int orow = q0 + r + (lhi << 3);
      out[(bT + orow) * H_ + nt * 16 + llo] = O[nt][r] / l[r];
    }
  }
}

// ---------------------------------------------------------------------------
extern "C" void kernel_launch(void* const* d_in, const int* in_sizes, int n_in,
                              void* d_out, int out_size, void* d_ws, size_t ws_size,
                              hipStream_t stream) {
  const float* x  = (const float*)d_in[0];
  const float* Wq = (const float*)d_in[1];
  const float* Wk = (const float*)d_in[2];
  const float* Wv = (const float*)d_in[3];
  float* out = (float*)d_out;

  _Float16* hq = (_Float16*)d_ws;
  _Float16* hk = hq + (size_t)B_ * T_ * H_;
  _Float16* hv = hk + (size_t)B_ * T_ * H_;
  _Float16* wt = hv + (size_t)B_ * T_ * H_;

  prep_w<<<dim3((3 * H_ * D_) / 256), dim3(256), 0, stream>>>(Wq, Wk, Wv, wt);
  proj_qkv<<<dim3((B_ * T_) / 16), dim3(384), 0, stream>>>(x, wt, hq, hk, hv);
  flash_attn<<<dim3(T_ / 64, B_), dim3(128), 0, stream>>>(hq, hk, hv, out);
}
